// BidirectionalMambaBlock_13511967113260
// MI455X (gfx1250) — matrix operand order repaired and checked by reading, not compiled, not run on hardware
//
#include <hip/hip_runtime.h>
#include <hip/hip_bf16.h>
#include <stdint.h>

typedef __attribute__((ext_vector_type(16))) __bf16 v16bf;
typedef __attribute__((ext_vector_type(8)))  float  v8f;
typedef __attribute__((ext_vector_type(4)))  unsigned int u32x4;
typedef __attribute__((ext_vector_type(8)))  unsigned int u32x8;

#define B_SZ   4
#define L_SZ   2048
#define DM     256
#define DI     512
#define DS     16
#define DTR    16
#define NROW   (B_SZ * L_SZ)     /* 8192 */
#define CHUNK  32
#define NCHUNK (L_SZ / CHUNK)    /* 64 */
#define DBL_LD 64                /* padded row stride of dbl */

__device__ __forceinline__ unsigned short f2bf(float f) {
  union { float f; unsigned int u; } x; x.f = f;
  unsigned int r = x.u + 0x7fffu + ((x.u >> 16) & 1u);
  return (unsigned short)(r >> 16);
}
__device__ __forceinline__ float sigm(float v) { return 1.f / (1.f + __expf(-v)); }

// ---------------------------------------------------------------- f32 -> bf16
__global__ void k_cvt_bf16(const float* __restrict__ in, unsigned short* __restrict__ out, int n) {
  int i = blockIdx.x * blockDim.x + threadIdx.x;
  if (i < n) out[i] = f2bf(in[i]);
}

// ------------------------------------- f32 (K x N) -> bf16 transposed (NPad x K)
__global__ void k_cvt_t(const float* __restrict__ in, unsigned short* __restrict__ out,
                        int Kd, int Nd, int NdPad) {
  int i = blockIdx.x * blockDim.x + threadIdx.x;
  if (i >= NdPad * Kd) return;
  int n = i / Kd, k = i - n * Kd;
  out[i] = (n < Nd) ? f2bf(in[(size_t)k * Nd + n]) : (unsigned short)0;
}

// --------------------------------------------------- TDM: stage 16x64 A tile
// D# group0: count=1 | lds_addr | global_addr(57b) | type=2
// D# group1: data_size=2B, tile 64x16, tensor_dim0_stride = lda
__device__ __forceinline__ void tdm_load_a(unsigned ldsOff, const unsigned short* gptr,
                                           int lda) {
  unsigned long long ga = (unsigned long long)(uintptr_t)gptr;
  u32x4 g0;
  g0[0] = 1u;                                            // count=1, user mode
  g0[1] = ldsOff;                                        // lds_addr (bytes)
  g0[2] = (unsigned)ga;                                  // global_addr[31:0]
  g0[3] = ((unsigned)(ga >> 32) & 0x1ffffffu) | (2u << 30); // addr[56:32] | type=2
  u32x8 g1;
  g1[0] = 1u << 16;                                      // data_size=1 (2 bytes)
  g1[1] = 0xffffu << 16;                                 // tensor_dim0 lo (huge)
  g1[2] = 0x7fffu | (0xffffu << 16);                     // dim0 hi | dim1 lo
  g1[3] = 0x7fffu | (64u << 16);                         // dim1 hi | tile_dim0=64
  g1[4] = 16u;                                           // tile_dim1=16
  g1[5] = (unsigned)lda;                                 // tensor_dim0_stride lo32
  g1[6] = 0u;
  g1[7] = 0u;
  asm volatile("tensor_load_to_lds %0, %1" :: "s"(g0), "s"(g1) : "memory");
}

// ------------------------------------------------------------- WMMA bf16 GEMM
// One wave computes a 16x64 tile of C = A(MxK) * Bt^T, Bt stored N x K (bf16).
// A tiles (16x64 per step) double-buffered through LDS via the TDM; B fragments
// software-pipelined one 32-K half ahead so WMMAs overlap L2 latency.
// mode: 0 = store f32; 1 = bias+relu, store bf16; 2 = bias, store f32.
union Frag { uint4 u[2]; v16bf v; unsigned short s[16]; };

__device__ __forceinline__ void loadB4(Frag* b, const unsigned short* __restrict__ Bt,
                                       int ldb, int colBase, int kofs) {
  #pragma unroll
  for (int i = 0; i < 4; ++i) {
    const uint4* bp = (const uint4*)(Bt + (size_t)(colBase + i * 16) * ldb + kofs);
    b[i].u[0] = bp[0];
    b[i].u[1] = bp[2];
  }
}

__global__ void k_gemm_bf16(
    const unsigned short* __restrict__ A, int lda,
    const unsigned short* __restrict__ Bt, int ldb,
    float* __restrict__ Cf, unsigned short* __restrict__ Cb,
    const float* __restrict__ bias,
    int M, int N, int K, int mode, int revA)
{
  __shared__ __align__(16) unsigned short As[8][2][16 * 64];  // 8 waves x 2 bufs x 2KB
  const int wave = threadIdx.x >> 5;
  const int lane = threadIdx.x & 31;
  const int ntiles = N >> 6;                  // 64-wide N tiles
  const int tile = blockIdx.x * 8 + wave;
  if (tile >= (M >> 4) * ntiles) return;
  const int tm = tile / ntiles, tn = tile - tm * ntiles;

  // Physical A row-block start (reverse direction: flipped block is contiguous)
  int grow = tm * 16;
  int physRow0 = grow;
  if (revA) { int b = grow >> 11, l0 = grow & (L_SZ - 1); physRow0 = b * L_SZ + (L_SZ - 16 - l0); }

  const int khalfA = (lane >> 4) * 8;   // A frag: K halves {0..7,16..23}/{8..15,24..31}
  const int kgrpB  = (lane >> 4) * 8;   // B frag: K halves {0..7,16..23}/{8..15,24..31}
  const int r      = lane & 15;
  const int ldsRow = revA ? (15 - r) : r;
  const unsigned ldsBase = (unsigned)wave * 4096u;
  const int colBase = tn * 64 + r;

  v8f acc[4];
  #pragma unroll
  for (int i = 0; i < 4; ++i) acc[i] = (v8f){0,0,0,0,0,0,0,0};

  const unsigned short* Ab = A + (size_t)physRow0 * lda;
  tdm_load_a(ldsBase, Ab, lda);               // prefetch kb=0..63 into buf 0
  Frag bA[4], bB[4];
  loadB4(bA, Bt, ldb, colBase, 0 + kgrpB);    // first 32-K half fragments
  int buf = 0;
  for (int kb = 0; kb < K; kb += 64) {
    const bool more = (kb + 64) < K;
    if (more) tdm_load_a(ldsBase + (unsigned)(buf ^ 1) * 2048u, Ab + kb + 64, lda);
    loadB4(bB, Bt, ldb, colBase, kb + 32 + kgrpB);   // second half (overlaps wmma)
    if (more) __builtin_amdgcn_s_wait_tensorcnt(1);
    else      __builtin_amdgcn_s_wait_tensorcnt(0);
    asm volatile("" ::: "memory");            // fence LDS reads behind the wait

    Frag a0, a1;
    const uint4* ap0 = (const uint4*)&As[wave][buf][ldsRow * 64 + khalfA];
    a0.u[0] = ap0[0];
    a0.u[1] = ap0[2];                         // +16 elements
    #pragma unroll
    for (int i = 0; i < 4; ++i)
      acc[i] = __builtin_amdgcn_wmma_f32_16x16x32_bf16(false, a0.v, false, bA[i].v,
                                                       (short)0, acc[i], false, false);

    const uint4* ap1 = (const uint4*)&As[wave][buf][ldsRow * 64 + 32 + khalfA];
    a1.u[0] = ap1[0];
    a1.u[1] = ap1[2];
    if (more) loadB4(bA, Bt, ldb, colBase, kb + 64 + kgrpB);  // next iter first half
    #pragma unroll
    for (int i = 0; i < 4; ++i)
      acc[i] = __builtin_amdgcn_wmma_f32_16x16x32_bf16(false, a1.v, false, bB[i].v,
                                                       (short)0, acc[i], false, false);
    buf ^= 1;
  }

  const int m0 = tm * 16 + ((lane >> 4) * 8);
  #pragma unroll
  for (int i = 0; i < 4; ++i) {
    const int n = tn * 64 + i * 16 + r;
    const float bv = bias ? bias[n] : 0.f;
    #pragma unroll
    for (int v = 0; v < 8; ++v) {
      float val = acc[i][v] + bv;
      if (mode == 1) val = fmaxf(val, 0.f);
      const size_t o = (size_t)(m0 + v) * N + n;
      if (Cf) Cf[o] = val;
      if (Cb) Cb[o] = f2bf(val);
    }
  }
}

// ------------------------------------------------------- depthwise conv + silu
__global__ void k_conv_silu(const float* __restrict__ xz, const float* __restrict__ convw,
                            const float* __restrict__ convb,
                            float* __restrict__ xcf, unsigned short* __restrict__ xcb)
{
  int i = blockIdx.x * blockDim.x + threadIdx.x;
  if (i >= NROW * DI) return;
  int row = i >> 9, d = i & 511;
  int l = row & (L_SZ - 1);
  float cur  = xz[(size_t)row * (2 * DI) + d];
  float prev = (l > 0) ? xz[(size_t)(row - 1) * (2 * DI) + d] : 0.f;
  float v = prev * convw[d * 2 + 0] + cur * convw[d * 2 + 1] + convb[d];
  float s = v * sigm(v);
  xcf[i] = s; xcb[i] = f2bf(s);
}

// ------------------------------------------------- dt projection + softplus
__global__ void k_dtproj(const float* __restrict__ dbl, const float* __restrict__ wdt,
                         const float* __restrict__ bdt, float* __restrict__ dt)
{
  int i = blockIdx.x * blockDim.x + threadIdx.x;
  if (i >= NROW * DI) return;
  int row = i >> 9, d = i & 511;
  float acc = bdt[d];
  const float* dr = dbl + (size_t)row * DBL_LD;
  #pragma unroll
  for (int r = 0; r < DTR; ++r) acc += dr[r] * wdt[r * DI + d];
  dt[i] = (acc > 20.f) ? acc : __logf(1.f + __expf(acc));
}

// --------------------------------------------- chunked scan pass 1 (local)
__global__ void k_scan_chunk(const float* __restrict__ dt, const float* __restrict__ xcf,
                             const float* __restrict__ dbl, const float* __restrict__ Alog,
                             float* __restrict__ hend, float* __restrict__ pprod)
{
  int gw = (blockIdx.x * blockDim.x + threadIdx.x) >> 5;
  int lane = threadIdx.x & 31;
  int dgrp = gw & 15; int t = gw >> 4;
  int chunk = t & (NCHUNK - 1); int b = t >> 6;
  if (b >= B_SZ) return;
  int d = dgrp * 32 + lane;
  float A[DS], h[DS], P[DS];
  #pragma unroll
  for (int n = 0; n < DS; ++n) { A[n] = -__expf(Alog[d * DS + n]); h[n] = 0.f; P[n] = 1.f; }
  int l0 = chunk * CHUNK;
  for (int l = l0; l < l0 + CHUNK; ++l) {
    int row = b * L_SZ + l;
    float dtv = dt[(size_t)row * DI + d];
    float xv  = xcf[(size_t)row * DI + d];
    const float* Br = dbl + (size_t)row * DBL_LD + DTR;
    float du = dtv * xv;
    #pragma unroll
    for (int n = 0; n < DS; ++n) {
      float dA = __expf(dtv * A[n]);
      h[n] = dA * h[n] + du * Br[n];
      P[n] *= dA;
    }
  }
  size_t o = (size_t)((b * NCHUNK + chunk) * DI + d) * DS;
  #pragma unroll
  for (int n = 0; n < DS; ++n) { hend[o + n] = h[n]; pprod[o + n] = P[n]; }
}

// ------------------------------------------ chunked scan pass 2 (carry, short)
__global__ void k_scan_carry(const float* __restrict__ hend, const float* __restrict__ pprod,
                             float* __restrict__ hin)
{
  int gw = (blockIdx.x * blockDim.x + threadIdx.x) >> 5;
  int lane = threadIdx.x & 31;
  int dgrp = gw & 15; int b = gw >> 4;
  if (b >= B_SZ) return;
  int d = dgrp * 32 + lane;
  float h[DS];
  #pragma unroll
  for (int n = 0; n < DS; ++n) h[n] = 0.f;
  for (int c = 0; c < NCHUNK; ++c) {
    size_t o = (size_t)((b * NCHUNK + c) * DI + d) * DS;
    #pragma unroll
    for (int n = 0; n < DS; ++n) {
      hin[o + n] = h[n];
      h[n] = pprod[o + n] * h[n] + hend[o + n];
    }
  }
}

// -------------------------------- chunked scan pass 3 (replay + C·h, gate)
__global__ void k_scan_out(const float* __restrict__ dt, const float* __restrict__ xcf,
                           const float* __restrict__ dbl, const float* __restrict__ Alog,
                           const float* __restrict__ Dp, const float* __restrict__ xz,
                           const float* __restrict__ hin, unsigned short* __restrict__ ybf)
{
  int gw = (blockIdx.x * blockDim.x + threadIdx.x) >> 5;
  int lane = threadIdx.x & 31;
  int dgrp = gw & 15; int t = gw >> 4;
  int chunk = t & (NCHUNK - 1); int b = t >> 6;
  if (b >= B_SZ) return;
  int d = dgrp * 32 + lane;
  float A[DS], h[DS];
  size_t o = (size_t)((b * NCHUNK + chunk) * DI + d) * DS;
  #pragma unroll
  for (int n = 0; n < DS; ++n) { A[n] = -__expf(Alog[d * DS + n]); h[n] = hin[o + n]; }
  float Dd = Dp[d];
  int l0 = chunk * CHUNK;
  for (int l = l0; l < l0 + CHUNK; ++l) {
    int row = b * L_SZ + l;
    float dtv = dt[(size_t)row * DI + d];
    float xv  = xcf[(size_t)row * DI + d];
    const float* Br = dbl + (size_t)row * DBL_LD + DTR;
    const float* Cr = dbl + (size_t)row * DBL_LD + DTR + DS;
    float du = dtv * xv;
    float y = 0.f;
    #pragma unroll
    for (int n = 0; n < DS; ++n) {
      float dA = __expf(dtv * A[n]);
      h[n] = dA * h[n] + du * Br[n];
      y += h[n] * Cr[n];
    }
    y += Dd * xv;
    float z = xz[(size_t)row * (2 * DI) + DI + d];
    y *= z * sigm(z);
    ybf[(size_t)row * DI + d] = f2bf(y);
  }
}

// ---------------------------------------------------------------- layernorms
__global__ void k_ln1(const float* __restrict__ x, const float* __restrict__ t1,
                      const float* __restrict__ t2, const float* __restrict__ g,
                      const float* __restrict__ bb, float* __restrict__ y3f,
                      unsigned short* __restrict__ y3b)
{
  __shared__ float red[256];
  int row = blockIdx.x, t = threadIdx.x;
  int b = row >> 11, l = row & (L_SZ - 1);
  size_t o  = (size_t)row * DM + t;
  size_t o2 = (size_t)(b * L_SZ + (L_SZ - 1 - l)) * DM + t;
  float v = x[o] + t1[o] + t2[o2];
  red[t] = v; __syncthreads();
  for (int s = 128; s > 0; s >>= 1) { if (t < s) red[t] += red[t + s]; __syncthreads(); }
  float mu = red[0] / DM; __syncthreads();
  float dv = v - mu;
  red[t] = dv * dv; __syncthreads();
  for (int s = 128; s > 0; s >>= 1) { if (t < s) red[t] += red[t + s]; __syncthreads(); }
  float r = rsqrtf(red[0] / DM + 1e-5f);
  float out = dv * r * g[t] + bb[t];
  y3f[o] = out; y3b[o] = f2bf(out);
}

__global__ void k_ln2(const float* __restrict__ cc, const float* __restrict__ y3f,
                      const float* __restrict__ g, const float* __restrict__ bb,
                      float* __restrict__ out)
{
  __shared__ float red[256];
  int row = blockIdx.x, t = threadIdx.x;
  size_t o = (size_t)row * DM + t;
  float v = cc[o] + y3f[o];
  red[t] = v; __syncthreads();
  for (int s = 128; s > 0; s >>= 1) { if (t < s) red[t] += red[t + s]; __syncthreads(); }
  float mu = red[0] / DM; __syncthreads();
  float dv = v - mu;
  red[t] = dv * dv; __syncthreads();
  for (int s = 128; s > 0; s >>= 1) { if (t < s) red[t] += red[t + s]; __syncthreads(); }
  float r = rsqrtf(red[0] / DM + 1e-5f);
  out[o] = dv * r * g[t] + bb[t];
}

// ---------------------------------------------------------------------------
extern "C" void kernel_launch(void* const* d_in, const int* in_sizes, int n_in,
                              void* d_out, int out_size, void* d_ws, size_t ws_size,
                              hipStream_t stream)
{
  const float* x = (const float*)d_in[0];
  const float* win[2]   = { (const float*)d_in[1],  (const float*)d_in[10] };
  const float* convw[2] = { (const float*)d_in[2],  (const float*)d_in[11] };
  const float* convb[2] = { (const float*)d_in[3],  (const float*)d_in[12] };
  const float* wx[2]    = { (const float*)d_in[4],  (const float*)d_in[13] };
  const float* wdt[2]   = { (const float*)d_in[5],  (const float*)d_in[14] };
  const float* bdt[2]   = { (const float*)d_in[6],  (const float*)d_in[15] };
  const float* Alog[2]  = { (const float*)d_in[7],  (const float*)d_in[16] };
  const float* Dp[2]    = { (const float*)d_in[8],  (const float*)d_in[17] };
  const float* wout[2]  = { (const float*)d_in[9],  (const float*)d_in[18] };
  const float* ln_g = (const float*)d_in[19];
  const float* ln_b = (const float*)d_in[20];
  const float* w1   = (const float*)d_in[21];
  const float* b1   = (const float*)d_in[22];
  const float* w3   = (const float*)d_in[23];
  const float* b3   = (const float*)d_in[24];

  char* ws = (char*)d_ws; size_t off = 0;
  auto alloc = [&](size_t bytes) -> void* {
    void* p = ws + off; off += (bytes + 255) & ~(size_t)255; return p;
  };

  unsigned short* xbf = (unsigned short*)alloc((size_t)NROW * DM * 2);
  unsigned short *winT[2], *wxT[2], *woutT[2];
  float *xz[2], *xcf[2], *dbl[2], *dtb[2], *hend[2], *pp[2], *hin[2];
  unsigned short *xcb[2], *ybf[2];
  for (int s = 0; s < 2; ++s) {
    winT[s]  = (unsigned short*)alloc((size_t)(2 * DI) * DM * 2);  // 1024 x 256
    wxT[s]   = (unsigned short*)alloc((size_t)DBL_LD * DI * 2);    // 64 x 512 (padded)
    woutT[s] = (unsigned short*)alloc((size_t)DM * DI * 2);        // 256 x 512
    xz[s]   = (float*)alloc((size_t)NROW * 2 * DI * 4);
    xcf[s]  = (float*)alloc((size_t)NROW * DI * 4);
    xcb[s]  = (unsigned short*)alloc((size_t)NROW * DI * 2);
    dbl[s]  = (float*)alloc((size_t)NROW * DBL_LD * 4);
    dtb[s]  = (float*)alloc((size_t)NROW * DI * 4);
    hend[s] = (float*)alloc((size_t)B_SZ * NCHUNK * DI * DS * 4);
    pp[s]   = (float*)alloc((size_t)B_SZ * NCHUNK * DI * DS * 4);
    hin[s]  = (float*)alloc((size_t)B_SZ * NCHUNK * DI * DS * 4);
    ybf[s]  = (unsigned short*)alloc((size_t)NROW * DI * 2);
  }
  unsigned short* w1bf = (unsigned short*)alloc((size_t)DM * DM * 2);
  unsigned short* w3bf = (unsigned short*)alloc((size_t)DM * DM * 2);
  float* t1  = (float*)alloc((size_t)NROW * DM * 4);
  float* t2  = (float*)alloc((size_t)NROW * DM * 4);
  float* y3f = (float*)alloc((size_t)NROW * DM * 4);
  unsigned short* y3b = (unsigned short*)alloc((size_t)NROW * DM * 2);
  unsigned short* abf = (unsigned short*)alloc((size_t)NROW * DM * 2);
  unsigned short* bbf = (unsigned short*)alloc((size_t)NROW * DM * 2);
  float* ccf = (float*)alloc((size_t)NROW * DM * 4);

  auto cvt = [&](const float* src, unsigned short* dst, int n) {
    k_cvt_bf16<<<(n + 255) / 256, 256, 0, stream>>>(src, dst, n);
  };
  auto cvtT = [&](const float* src, unsigned short* dst, int Kd, int Nd, int NdPad) {
    int n = NdPad * Kd;
    k_cvt_t<<<(n + 255) / 256, 256, 0, stream>>>(src, dst, Kd, Nd, NdPad);
  };
  auto gemm = [&](const unsigned short* A, int lda, const unsigned short* Bt, int ldb,
                  float* Cf, unsigned short* Cb, const float* bias,
                  int M, int N, int K, int mode, int rev) {
    int tiles = (M / 16) * (N / 64);
    k_gemm_bf16<<<(tiles + 7) / 8, 256, 0, stream>>>(A, lda, Bt, ldb,
                                                     Cf, Cb, bias, M, N, K, mode, rev);
  };

  cvt(x, xbf, NROW * DM);
  cvt(w1, w1bf, DM * DM);      // w1 is already (out x in) = N x K
  cvt(w3, w3bf, DM * DM);
  for (int s = 0; s < 2; ++s) {
    cvtT(win[s],  winT[s],  DM, 2 * DI, 2 * DI);
    cvtT(wx[s],   wxT[s],   DI, 48, DBL_LD);
    cvtT(wout[s], woutT[s], DI, DM, DM);
  }

  const int nElemDI = NROW * DI;
  const int waves1 = B_SZ * NCHUNK * (DI / 32);          // 4096 waves
  for (int s = 0; s < 2; ++s) {
    // xz = x @ win (reverse direction: flipped rows folded into TDM tile base)
    gemm(xbf, DM, winT[s], DM, xz[s], nullptr, nullptr, NROW, 2 * DI, DM, 0, s);
    k_conv_silu<<<(nElemDI + 255) / 256, 256, 0, stream>>>(xz[s], convw[s], convb[s],
                                                           xcf[s], xcb[s]);
    // dbl = xc @ wx (N padded to 64)
    gemm(xcb[s], DI, wxT[s], DI, dbl[s], nullptr, nullptr, NROW, DBL_LD, DI, 0, 0);
    k_dtproj<<<(nElemDI + 255) / 256, 256, 0, stream>>>(dbl[s], wdt[s], bdt[s], dtb[s]);
    // chunked selective scan (serial depth 2048 -> 32 + 64)
    k_scan_chunk<<<waves1 * 32 / 256, 256, 0, stream>>>(dtb[s], xcf[s], dbl[s], Alog[s],
                                                        hend[s], pp[s]);
    k_scan_carry<<<(B_SZ * (DI / 32) * 32 + 255) / 256, 256, 0, stream>>>(hend[s], pp[s], hin[s]);
    k_scan_out<<<waves1 * 32 / 256, 256, 0, stream>>>(dtb[s], xcf[s], dbl[s], Alog[s],
                                                      Dp[s], xz[s], hin[s], ybf[s]);
    // t = y @ wout
    gemm(ybf[s], DI, woutT[s], DI, (s == 0) ? t1 : t2, nullptr, nullptr,
         NROW, DM, DI, 0, 0);
  }

  // y3 = LN(x + y1 + flip(y2))
  k_ln1<<<NROW, 256, 0, stream>>>(x, t1, t2, ln_g, ln_b, y3f, y3b);
  // MLP: a = relu(y3 @ w1^T + b1); b = relu(a @ w3^T + b3); c = b @ w3^T + b3
  gemm(y3b, DM, w1bf, DM, nullptr, abf, b1, NROW, DM, DM, 1, 0);
  gemm(abf, DM, w3bf, DM, nullptr, bbf, b3, NROW, DM, DM, 1, 0);
  gemm(bbf, DM, w3bf, DM, ccf, nullptr, b3, NROW, DM, DM, 2, 0);
  // out = LN(c + y3)
  k_ln2<<<NROW, 256, 0, stream>>>(ccf, y3f, ln_g, ln_b, (float*)d_out);
  (void)in_sizes; (void)n_in; (void)out_size; (void)ws_size;
}
